// ECCN_37099927503187
// MI455X (gfx1250) — compile-verified, matrix operand order checked by reading, not run on hardware
//
#include <hip/hip_runtime.h>
#include <hip/hip_bf16.h>

typedef __attribute__((ext_vector_type(16))) _Float16 v16h;
typedef __attribute__((ext_vector_type(8)))  float    v8f;
typedef __attribute__((ext_vector_type(4)))  unsigned v4u;
typedef __attribute__((ext_vector_type(8)))  int      v8i;
typedef __attribute__((ext_vector_type(4)))  int      v4i;

#define NF 128
#define H  64
#define HH 4096   // H*H

// LDS carve for the persistent nnconv kernel (dynamic shared memory)
#define LDS_B_BYTES    (HH * 32 * 2)          // 262144: packed f16 B matrix
#define LDS_BIAS_OFF   LDS_B_BYTES            // 16384:  edge-MLP bias (f32)
#define LDS_XS_OFF     (LDS_B_BYTES + HH * 4) // 8 waves x 16x64 f32 gather tiles
#define LDS_TOTAL      (LDS_XS_OFF + 8 * 16 * H * 4)   // 311296 <= 320KB

// ---------------------------------------------------------------------------
// Pack edge_attr [E,32] f32 -> f16 in WMMA A-layout, 16 edges per block/wave.
// A 16x32 f16: lanes 0-15 hold M=lane, K=0..7 / K=16..23; lanes 16-31 hold
// M=lane-16, K=8..15 / K=24..31.
// ---------------------------------------------------------------------------
__global__ __launch_bounds__(32) void pack_a_kernel(const float* __restrict__ ea,
                                                    _Float16* __restrict__ pA) {
    int tile = blockIdx.x, lane = threadIdx.x;
    int m  = lane & 15;
    int kb = (lane < 16) ? 0 : 8;
    const float* row = ea + (size_t)(tile * 16 + m) * 32;
    v16h t;
#pragma unroll
    for (int i = 0; i < 8; i++) {
        t[i]     = (_Float16)row[kb + i];
        t[i + 8] = (_Float16)row[16 + kb + i];
    }
    *(v16h*)(pA + (size_t)tile * 512 + lane * 16) = t;
}

// ---------------------------------------------------------------------------
// Pack fce_w [32,4096] f32 -> f16 in WMMA B-layout, 16 columns per block/wave.
// ---------------------------------------------------------------------------
__global__ __launch_bounds__(32) void pack_b_kernel(const float* __restrict__ w,
                                                    _Float16* __restrict__ pB) {
    int t = blockIdx.x, lane = threadIdx.x;
    int n  = t * 16 + (lane & 15);
    int kb = (lane < 16) ? 0 : 8;
    v16h v;
#pragma unroll
    for (int i = 0; i < 8; i++) {
        v[i]     = (_Float16)w[(size_t)(kb + i) * HH + n];
        v[i + 8] = (_Float16)w[(size_t)(16 + kb + i) * HH + n];
    }
    *(v16h*)(pB + (size_t)t * 512 + lane * 16) = v;
}

// ---------------------------------------------------------------------------
// hx = x @ fc1_w + fc1_b        [N,128]@[128,64]
// ---------------------------------------------------------------------------
__global__ __launch_bounds__(256) void fc1_kernel(const float* __restrict__ x,
                                                  const float* __restrict__ w,
                                                  const float* __restrict__ b,
                                                  float* __restrict__ out) {
    __shared__ float xs[4 * NF];
    int node0 = blockIdx.x * 4;
    for (int i = threadIdx.x; i < 4 * NF; i += 256) xs[i] = x[(size_t)node0 * NF + i];
    __syncthreads();
    int ln = threadIdx.x >> 6, o = threadIdx.x & 63;
    float s = b[o];
#pragma unroll 8
    for (int k = 0; k < NF; k++) s += xs[ln * NF + k] * w[k * H + o];
    out[(size_t)(node0 + ln) * H + o] = s;
}

// ---------------------------------------------------------------------------
// agg = h @ root + bias          [N,64]@[64,64]  (pre-init for scatter-add)
// ---------------------------------------------------------------------------
__global__ __launch_bounds__(256) void init_agg_kernel(const float* __restrict__ h,
                                                       const float* __restrict__ root,
                                                       const float* __restrict__ bias,
                                                       float* __restrict__ agg) {
    __shared__ float hs[4 * H];
    int node0 = blockIdx.x * 4;
    for (int i = threadIdx.x; i < 4 * H; i += 256) hs[i] = h[(size_t)node0 * H + i];
    __syncthreads();
    int ln = threadIdx.x >> 6, o = threadIdx.x & 63;
    float s = bias[o];
#pragma unroll 8
    for (int k = 0; k < H; k++) s += hs[ln * H + k] * root[k * H + o];
    agg[(size_t)(node0 + ln) * H + o] = s;
}

// ---------------------------------------------------------------------------
// Fused NNConv message pass, persistent blocks.
//  - Packed B (256KB f16) DMA'd into LDS ONCE per block via the Tensor Data
//    Mover (tensor_load_to_lds + s_wait_tensorcnt), shared by 8 waves.
//  - Each wave processes 16-edge tiles in a grid-stride loop:
//      W = relu(A(16x32) x B + ebias) via 256 V_WMMA_F32_16X16X32_F16,
//      msg += x[src,h] * W folded in-register, atomicAdd scatter to agg[dst].
// ---------------------------------------------------------------------------
__global__ __launch_bounds__(256) void nnconv_kernel(const _Float16* __restrict__ pA,
                                                     const _Float16* __restrict__ pB,
                                                     const float* __restrict__ ebias,
                                                     const float* __restrict__ feats,
                                                     const int* __restrict__ srcIdx,
                                                     const int* __restrict__ dstIdx,
                                                     float* __restrict__ agg,
                                                     int ntiles) {
    extern __shared__ char smem[];
    _Float16* Bsh = (_Float16*)smem;
    float*    bsh = (float*)(smem + LDS_BIAS_OFF);

    int lane = threadIdx.x & 31;
    int wave = threadIdx.x >> 5;
    float* xsh = (float*)(smem + LDS_XS_OFF + wave * (16 * H * 4));

    // --- stage packed B into LDS with the TDM (wave 0 only) ---------------
    if (wave == 0) {
        unsigned long long ga = (unsigned long long)pB;
        unsigned lds_base = __builtin_amdgcn_groupstaticsize();   // dyn-LDS base
        v4u g0;
        g0[0] = 1u;                                   // count=1, user descriptor
        g0[1] = lds_base;                             // lds_addr (B at offset 0)
        g0[2] = (unsigned)(ga & 0xffffffffu);         // global_addr[31:0]
        g0[3] = (unsigned)((ga >> 32) & 0x01ffffffu)  // global_addr[56:32]
              | 0x80000000u;                          // type=2 ("image")
        v8i g1;
        g1[0] = (3 << 16);          // data_size = 8 bytes
        g1[1] = (int)0x80000000;    // tensor_dim0 = 32768 (bits 79:48)
        g1[2] = 0x00010000;         // tensor_dim1 = 1     (bits 111:80)
        g1[3] = (int)0x80000000;    // tile_dim0   = 32768 (bits 127:112)
        g1[4] = 1;                  // tile_dim1 = 1, tile_dim2 = 0
        g1[5] = 32768;              // tensor_dim0_stride (bits 207:160)
        g1[6] = 0;
        g1[7] = 0;
        v4i g2 = {0, 0, 0, 0};
        v4i g3 = {0, 0, 0, 0};
        v8i g4 = {0, 0, 0, 0, 0, 0, 0, 0};
        __builtin_amdgcn_tensor_load_to_lds(g0, g1, g2, g3, g4, 0);
        __builtin_amdgcn_s_wait_tensorcnt(0);
    }
    // --- stage edge-MLP bias (all threads) --------------------------------
    for (int i = threadIdx.x; i < HH; i += 256) bsh[i] = ebias[i];
    __syncthreads();

    int nlane = lane & 15;
    int mb    = (lane >> 4) << 3;   // 0 or 8: row base this half-wave owns

    for (int tile = blockIdx.x * 8 + wave; tile < ntiles; tile += gridDim.x * 8) {
        int m = lane & 15;
        int sreg = srcIdx[tile * 16 + m];
        int dreg = dstIdx[tile * 16 + m];
        // gather x[src] rows (16 x 64 floats) into this wave's LDS tile
        for (int r = 0; r < 16; r++) {
            int s = __shfl(sreg, r, 32);
            xsh[r * H + lane]      = feats[(size_t)s * H + lane];
            xsh[r * H + 32 + lane] = feats[(size_t)s * H + 32 + lane];
        }

        v16h a = *(const v16h*)(pA + (size_t)tile * 512 + lane * 16);
        v8f acc[4] = {v8f{}, v8f{}, v8f{}, v8f{}};

        for (int h = 0; h < H; h++) {
            float xv[8];
#pragma unroll
            for (int v = 0; v < 8; v++) xv[v] = xsh[(mb + v) * H + h];
#pragma unroll
            for (int ob = 0; ob < 4; ob++) {
                int blk = h * 4 + ob;   // flat col block c = h*64 + ob*16
                v16h bm = *(const v16h*)(Bsh + (size_t)blk * 512 + lane * 16);
                float cb = bsh[blk * 16 + nlane];
                v8f c = {cb, cb, cb, cb, cb, cb, cb, cb};
                v8f d = __builtin_amdgcn_wmma_f32_16x16x32_f16(
                    false, a, false, bm, (short)0, c, false, false);
#pragma unroll
                for (int v = 0; v < 8; v++) {
                    float r = d[v];
                    r = (r > 0.0f) ? r : 0.0f;     // relu -> single v_max_num
                    acc[ob][v] += r * xv[v];
                }
            }
        }
        // scatter-add messages to target nodes
#pragma unroll
        for (int ob = 0; ob < 4; ob++) {
#pragma unroll
            for (int v = 0; v < 8; v++) {
                int dn = __shfl(dreg, mb + v, 32);
                atomicAdd(&agg[(size_t)dn * H + ob * 16 + nlane], acc[ob][v]);
            }
        }
    }
}

// ---------------------------------------------------------------------------
// Global sum / sum-of-squares reduction (graph norm statistics).
// ---------------------------------------------------------------------------
__global__ __launch_bounds__(256) void reduce_kernel(const float* __restrict__ v,
                                                     int n, float* __restrict__ sums) {
    float s = 0.f, ss = 0.f;
    for (int i = blockIdx.x * blockDim.x + threadIdx.x; i < n; i += gridDim.x * blockDim.x) {
        float t = v[i];
        s += t; ss += t * t;
    }
#pragma unroll
    for (int off = 16; off; off >>= 1) {
        s  += __shfl_xor(s, off, 32);
        ss += __shfl_xor(ss, off, 32);
    }
    __shared__ float as[8], bs[8];
    int wid = threadIdx.x >> 5;
    if ((threadIdx.x & 31) == 0) { as[wid] = s; bs[wid] = ss; }
    __syncthreads();
    if (threadIdx.x == 0) {
        float ta = 0.f, tb = 0.f;
        for (int i = 0; i < 8; i++) { ta += as[i]; tb += bs[i]; }
        atomicAdd(&sums[0], ta);
        atomicAdd(&sums[1], tb);
    }
}

__global__ __launch_bounds__(32) void zero4_kernel(float* s) {
    if (threadIdx.x < 4) s[threadIdx.x] = 0.f;
}

// out = relu( w[c] * (in - mu)*rstd + b[c] )
__global__ __launch_bounds__(256) void gnorm_apply_kernel(const float* __restrict__ in,
                                                          const float* __restrict__ sums,
                                                          const float* __restrict__ w,
                                                          const float* __restrict__ b,
                                                          float* __restrict__ out, int n) {
    int i = blockIdx.x * blockDim.x + threadIdx.x;
    if (i >= n) return;
    float cnt  = (float)n;
    float mu   = sums[0] / cnt;
    float var  = sums[1] / cnt - mu * mu;
    float rstd = rsqrtf(var + 1e-5f);
    int o = i & 63;
    float val = w[o] * ((in[i] - mu) * rstd) + b[o];
    out[i] = (val > 0.0f) ? val : 0.0f;
}

// out[n] = h[n,:] . fc2_w + fc2_b
__global__ __launch_bounds__(256) void fc2_kernel(const float* __restrict__ h,
                                                  const float* __restrict__ w,
                                                  const float* __restrict__ b,
                                                  float* __restrict__ out, int n) {
    int i = blockIdx.x * blockDim.x + threadIdx.x;
    if (i >= n) return;
    float s = b[0];
    const float* row = h + (size_t)i * H;
#pragma unroll 8
    for (int k = 0; k < H; k++) s += row[k] * w[k];
    out[i] = s;
}

// ---------------------------------------------------------------------------
extern "C" void kernel_launch(void* const* d_in, const int* in_sizes, int n_in,
                              void* d_out, int out_size, void* d_ws, size_t ws_size,
                              hipStream_t stream) {
    const float* x      = (const float*)d_in[0];
    const int*   ei     = (const int*)d_in[1];
    const float* ea     = (const float*)d_in[2];
    const float* fc1_w  = (const float*)d_in[3];
    const float* fc1_b  = (const float*)d_in[4];
    const float* fce1_w = (const float*)d_in[5];
    const float* fce1_b = (const float*)d_in[6];
    const float* root1  = (const float*)d_in[7];
    const float* bias1  = (const float*)d_in[8];
    const float* gn1_w  = (const float*)d_in[9];
    const float* gn1_b  = (const float*)d_in[10];
    const float* fce2_w = (const float*)d_in[11];
    const float* fce2_b = (const float*)d_in[12];
    const float* root2  = (const float*)d_in[13];
    const float* bias2  = (const float*)d_in[14];
    const float* gn2_w  = (const float*)d_in[15];
    const float* gn2_b  = (const float*)d_in[16];
    const float* fc2_w  = (const float*)d_in[17];
    const float* fc2_b  = (const float*)d_in[18];

    const int N = in_sizes[0] / NF;   // 20000
    const int E = in_sizes[2] / 32;   // 50000
    const int ntiles = E / 16;        // 3125
    const int* src = ei;
    const int* dst = ei + E;

    // workspace carve-up
    char* w = (char*)d_ws;
    size_t off = 0;
    _Float16* pA  = (_Float16*)(w + off); off += (size_t)E * 32 * 2; off = (off + 255) & ~(size_t)255;
    _Float16* pB1 = (_Float16*)(w + off); off += (size_t)HH * 32 * 2; off = (off + 255) & ~(size_t)255;
    _Float16* pB2 = (_Float16*)(w + off); off += (size_t)HH * 32 * 2; off = (off + 255) & ~(size_t)255;
    float* B0   = (float*)(w + off); off += (size_t)N * H * 4; off = (off + 255) & ~(size_t)255;
    float* B1   = (float*)(w + off); off += (size_t)N * H * 4; off = (off + 255) & ~(size_t)255;
    float* sums = (float*)(w + off);

    const int nnconv_blocks = 128;    // persistent; 8 waves each

    // pack f16 operands (A shared by both layers)
    pack_a_kernel<<<ntiles, 32, 0, stream>>>(ea, pA);
    pack_b_kernel<<<HH / 16, 32, 0, stream>>>(fce1_w, pB1);
    pack_b_kernel<<<HH / 16, 32, 0, stream>>>(fce2_w, pB2);
    zero4_kernel<<<1, 32, 0, stream>>>(sums);

    // hx = x @ fc1 + b
    fc1_kernel<<<N / 4, 256, 0, stream>>>(x, fc1_w, fc1_b, B0);

    // ---- layer 1 ----
    init_agg_kernel<<<N / 4, 256, 0, stream>>>(B0, root1, bias1, B1);
    nnconv_kernel<<<nnconv_blocks, 256, LDS_TOTAL, stream>>>(pA, pB1, fce1_b, B0, src, dst, B1, ntiles);
    reduce_kernel<<<512, 256, 0, stream>>>(B1, N * H, sums);
    gnorm_apply_kernel<<<(N * H + 255) / 256, 256, 0, stream>>>(B1, sums, gn1_w, gn1_b, B0, N * H);

    // ---- layer 2 ----
    init_agg_kernel<<<N / 4, 256, 0, stream>>>(B0, root2, bias2, B1);
    nnconv_kernel<<<nnconv_blocks, 256, LDS_TOTAL, stream>>>(pA, pB2, fce2_b, B0, src, dst, B1, ntiles);
    reduce_kernel<<<512, 256, 0, stream>>>(B1, N * H, sums + 2);
    gnorm_apply_kernel<<<(N * H + 255) / 256, 256, 0, stream>>>(B1, sums + 2, gn2_w, gn2_b, B0, N * H);

    // readout
    fc2_kernel<<<(N + 255) / 256, 256, 0, stream>>>(B0, fc2_w, fc2_b, (float*)d_out, N);
}
